// MemMambaBlock_12000138625497
// MI455X (gfx1250) — matrix operand order repaired and checked by reading, not compiled, not run on hardware
//
#include <hip/hip_runtime.h>
#include <hip/hip_bf16.h>
#include <math.h>

typedef __bf16 bf16_t;
typedef __attribute__((ext_vector_type(16))) __bf16 v16bf;
typedef __attribute__((ext_vector_type(8)))  __bf16 v8bf;
typedef __attribute__((ext_vector_type(8)))  float  v8f;

#define D_MODEL   1024
#define D_STATE   128
#define D_CONV    4
#define HEADDIM   64
#define D_INNER   2048
#define NHEADS    32
#define CONV_DIM  2304      // D_INNER + 2*D_STATE
#define D_IN_PROJ 4384      // 2*D_INNER + 2*D_STATE + NHEADS
#define POOL_N    50
#define SDIM      64
#define HID       256
#define BATCH     4
#define TLEN      2048
#define BT        (BATCH*TLEN)   // 8192
#define EPSF      1e-5f
#define TAU1F     0.5f
#define TAU2F     0.3f

// ---------------------------------------------------------------------------
// Generic f32 -> bf16 conversion
// ---------------------------------------------------------------------------
__global__ void f32_to_bf16_kernel(const float* __restrict__ in,
                                   bf16_t* __restrict__ out, long long n) {
  long long i = (long long)blockIdx.x * blockDim.x + threadIdx.x;
  if (i < n) out[i] = (bf16_t)in[i];
}

// ---------------------------------------------------------------------------
// RMSNorm (row length = len), bf16 output (GEMM operand)
// ---------------------------------------------------------------------------
__global__ __launch_bounds__(256) void rmsnorm_bf16_kernel(
    const float* __restrict__ x, const float* __restrict__ w,
    bf16_t* __restrict__ obf, int len) {
  const int row = blockIdx.x;
  const float* xr = x + (size_t)row * len;
  __shared__ float red[256];
  float s = 0.f;
  for (int i = threadIdx.x; i < len; i += 256) { float v = xr[i]; s += v * v; }
  red[threadIdx.x] = s; __syncthreads();
  for (int st = 128; st > 0; st >>= 1) {
    if (threadIdx.x < st) red[threadIdx.x] += red[threadIdx.x + st];
    __syncthreads();
  }
  const float inv = rsqrtf(red[0] / (float)len + EPSF);
  for (int i = threadIdx.x; i < len; i += 256)
    obf[(size_t)row * len + i] = (bf16_t)(xr[i] * inv * w[i]);
}

// ---------------------------------------------------------------------------
// bf16 WMMA GEMM: C[M,N] = A[M,K] * W[N,K]^T  (row-major A and W)
// wave computes a 32x32 tile via 4x v_wmma_f32_16x16x32_bf16 per K-step.
// Requires M%32==0, N%32==0, K%32==0. relu!=0 applies max(x,0) epilogue.
// ---------------------------------------------------------------------------
__global__ __launch_bounds__(256) void gemm_bf16_wmma_kernel(
    const bf16_t* __restrict__ A, const bf16_t* __restrict__ W,
    float* __restrict__ C, int M, int N, int K, int relu) {
  const int lane   = threadIdx.x & 31;
  const int wave   = threadIdx.x >> 5;
  const int tilesN = N >> 5;
  const int tilesM = M >> 5;
  const int tile   = blockIdx.x * 8 + wave;
  if (tile >= tilesM * tilesN) return;
  const int tm = tile / tilesN, tn = tile - tm * tilesN;
  const int m0 = tm << 5, n0 = tn << 5;
  const int half = lane >> 4;
  const int l15  = lane & 15;

  v8f acc00 = {}, acc01 = {}, acc10 = {}, acc11 = {};

  // A fragment (16x32 bf16): lane m=l15, K packing: e<8 -> k=half*8+e,
  // e>=8 -> k=16+half*8+(e-8)  => two contiguous 16B chunks at +0 and +16.
  const bf16_t* pa0 = A + (size_t)(m0 + l15) * K + half * 8;
  const bf16_t* pa1 = pa0 + (size_t)16 * K;
  // B fragment (32x16 bf16): lane col n=l15, same K packing as A:
  // e<8 -> k=half*8+e, e>=8 -> k=16+half*8+(e-8) along K of W row n.
  const bf16_t* pw0 = W + (size_t)(n0 + l15) * K + half * 8;
  const bf16_t* pw1 = pw0 + (size_t)16 * K;

  for (int kb = 0; kb < K; kb += 32) {
    union { v16bf v; v8bf h[2]; } ua0, ua1, ub0, ub1;
    ua0.h[0] = *(const v8bf*)(pa0);  ua0.h[1] = *(const v8bf*)(pa0 + 16);
    ua1.h[0] = *(const v8bf*)(pa1);  ua1.h[1] = *(const v8bf*)(pa1 + 16);
    ub0.h[0] = *(const v8bf*)(pw0);  ub0.h[1] = *(const v8bf*)(pw0 + 16);
    ub1.h[0] = *(const v8bf*)(pw1);  ub1.h[1] = *(const v8bf*)(pw1 + 16);
    __builtin_prefetch(pa0 + 64, 0, 1);   // global_prefetch_b8 (next K tiles)
    __builtin_prefetch(pw0 + 64, 0, 1);
    acc00 = __builtin_amdgcn_wmma_f32_16x16x32_bf16(false, ua0.v, false, ub0.v,
                                                    (short)0, acc00, false, false);
    acc01 = __builtin_amdgcn_wmma_f32_16x16x32_bf16(false, ua0.v, false, ub1.v,
                                                    (short)0, acc01, false, false);
    acc10 = __builtin_amdgcn_wmma_f32_16x16x32_bf16(false, ua1.v, false, ub0.v,
                                                    (short)0, acc10, false, false);
    acc11 = __builtin_amdgcn_wmma_f32_16x16x32_bf16(false, ua1.v, false, ub1.v,
                                                    (short)0, acc11, false, false);
    pa0 += 32; pa1 += 32; pw0 += 32; pw1 += 32;
  }

  // C/D layout: VGPR r -> M = half*8 + r (within 16x16 tile), N = l15.
  const int cn0 = n0 + l15, cn1 = cn0 + 16;
#pragma unroll
  for (int r = 0; r < 8; ++r) {
    const int mr = m0 + half * 8 + r;
    float v00 = acc00[r], v01 = acc01[r], v10 = acc10[r], v11 = acc11[r];
    if (relu) {
      v00 = fmaxf(v00, 0.f); v01 = fmaxf(v01, 0.f);
      v10 = fmaxf(v10, 0.f); v11 = fmaxf(v11, 0.f);
    }
    C[(size_t)mr * N + cn0]        = v00;
    C[(size_t)mr * N + cn1]        = v01;
    C[(size_t)(mr + 16) * N + cn0] = v10;
    C[(size_t)(mr + 16) * N + cn1] = v11;
  }
}

// ---------------------------------------------------------------------------
// Depthwise causal conv (window 4) + bias + SiLU over zxbcdt[:, 2048:4352]
// ---------------------------------------------------------------------------
__global__ void conv_silu_kernel(const float* __restrict__ zxbcdt,
                                 const float* __restrict__ conv_w,
                                 const float* __restrict__ conv_b,
                                 float* __restrict__ xBC) {
  long long idx = (long long)blockIdx.x * blockDim.x + threadIdx.x;
  if (idx >= (long long)BT * CONV_DIM) return;
  const int c  = (int)(idx % CONV_DIM);
  const int bt = (int)(idx / CONV_DIM);
  const int b  = bt / TLEN, t = bt % TLEN;
  float acc = conv_b[c];
#pragma unroll
  for (int j = 0; j < D_CONV; ++j) {
    const int tt = t - (D_CONV - 1) + j;
    if (tt >= 0)
      acc += zxbcdt[(size_t)(b * TLEN + tt) * D_IN_PROJ + D_INNER + c] * conv_w[c * D_CONV + j];
  }
  xBC[idx] = acc / (1.f + expf(-acc));   // silu
}

// ---------------------------------------------------------------------------
// dt = softplus(raw + dt_bias), dA = exp(dt * (-exp(A_log)))
// ---------------------------------------------------------------------------
__global__ void dt_kernel(const float* __restrict__ zxbcdt,
                          const float* __restrict__ dt_bias,
                          const float* __restrict__ A_log,
                          float* __restrict__ dt, float* __restrict__ dA) {
  long long idx = (long long)blockIdx.x * blockDim.x + threadIdx.x;
  if (idx >= (long long)BT * NHEADS) return;
  const int h  = (int)(idx % NHEADS);
  const int bt = (int)(idx / NHEADS);
  float v = zxbcdt[(size_t)bt * D_IN_PROJ + D_INNER + CONV_DIM + h] + dt_bias[h];
  float sp = (v > 20.f) ? v : log1pf(expf(v));
  dt[idx] = sp;
  dA[idx] = expf(sp * (-expf(A_log[h])));
}

// ---------------------------------------------------------------------------
// SSM scan: one block per (b, head). State h[64][128] in registers
// (thread t: p = t>>2, n in [32*(t&3), 32*(t&3)+32)). B/C/x via LDS.
// ---------------------------------------------------------------------------
__global__ __launch_bounds__(256) void ssm_scan_kernel(
    const float* __restrict__ xBC, const float* __restrict__ dt,
    const float* __restrict__ dA, const float* __restrict__ D_param,
    float* __restrict__ yssm) {
  const int b    = blockIdx.x / NHEADS;
  const int head = blockIdx.x % NHEADS;
  const int tid  = threadIdx.x;
  const int p    = tid >> 2;
  const int n0   = (tid & 3) * 32;

  __shared__ float sB[D_STATE];
  __shared__ float sC[D_STATE];
  __shared__ float sX[HEADDIM];
  __shared__ float sScal[2];

  float h[32];
#pragma unroll
  for (int i = 0; i < 32; ++i) h[i] = 0.f;

  const float Dp = D_param[head];

  for (int t = 0; t < TLEN; ++t) {
    const size_t row = (size_t)b * TLEN + t;
    if (tid < 128)       sB[tid]       = xBC[row * CONV_DIM + D_INNER + tid];
    else                 sC[tid - 128] = xBC[row * CONV_DIM + D_INNER + D_STATE + (tid - 128)];
    if (tid < 64)        sX[tid]       = xBC[row * CONV_DIM + head * HEADDIM + tid];
    if (tid == 0) { sScal[0] = dA[row * NHEADS + head]; sScal[1] = dt[row * NHEADS + head]; }
    __syncthreads();

    const float dAv = sScal[0];
    const float xv  = sX[p];
    const float dtx = sScal[1] * xv;
    float yp = 0.f;
#pragma unroll
    for (int i = 0; i < 32; ++i) {
      h[i] = h[i] * dAv + dtx * sB[n0 + i];
      yp  += h[i] * sC[n0 + i];
    }
    // reduce over the 4 n-chunks (adjacent lanes)
    yp += __shfl_xor(yp, 1, 32);
    yp += __shfl_xor(yp, 2, 32);
    if ((tid & 3) == 0)
      yssm[row * D_INNER + head * HEADDIM + p] = yp + Dp * xv;
    __syncthreads();
  }
}

// ---------------------------------------------------------------------------
// Gated RMSNorm: v = yssm * silu(z);  out_bf16 = rmsnorm(v) * gnorm_w
// ---------------------------------------------------------------------------
__global__ __launch_bounds__(256) void gated_rmsnorm_kernel(
    const float* __restrict__ yssm, const float* __restrict__ zxbcdt,
    const float* __restrict__ gw, bf16_t* __restrict__ out) {
  const int row = blockIdx.x;
  const float* yr = yssm + (size_t)row * D_INNER;
  const float* zr = zxbcdt + (size_t)row * D_IN_PROJ;   // z = cols [0,2048)
  __shared__ float red[256];
  float s = 0.f;
  for (int i = threadIdx.x; i < D_INNER; i += 256) {
    float z = zr[i];
    float v = yr[i] * (z / (1.f + expf(-z)));
    s += v * v;
  }
  red[threadIdx.x] = s; __syncthreads();
  for (int st = 128; st > 0; st >>= 1) {
    if (threadIdx.x < st) red[threadIdx.x] += red[threadIdx.x + st];
    __syncthreads();
  }
  const float inv = rsqrtf(red[0] / (float)D_INNER + EPSF);
  for (int i = threadIdx.x; i < D_INNER; i += 256) {
    float z = zr[i];
    float v = yr[i] * (z / (1.f + expf(-z)));
    out[(size_t)row * D_INNER + i] = (bf16_t)(v * inv * gw[i]);
  }
}

// ---------------------------------------------------------------------------
// scores = sigmoid(h_relu . w2), one thread per row
// ---------------------------------------------------------------------------
__global__ void score_kernel(const float* __restrict__ h,
                             const float* __restrict__ w2,
                             float* __restrict__ scores, int rows) {
  int row = blockIdx.x * blockDim.x + threadIdx.x;
  if (row >= rows) return;
  const float* hr = h + (size_t)row * HID;
  float acc = 0.f;
  for (int i = 0; i < HID; ++i) acc += hr[i] * w2[i];
  scores[row] = 1.f / (1.f + expf(-acc));
}

// ---------------------------------------------------------------------------
// per-batch mean of scores
// ---------------------------------------------------------------------------
__global__ __launch_bounds__(256) void mean_kernel(const float* __restrict__ scores,
                                                   float* __restrict__ ms) {
  const int b = blockIdx.x;
  __shared__ float red[256];
  float s = 0.f;
  for (int i = threadIdx.x; i < TLEN; i += 256) s += scores[b * TLEN + i];
  red[threadIdx.x] = s; __syncthreads();
  for (int st = 128; st > 0; st >>= 1) {
    if (threadIdx.x < st) red[threadIdx.x] += red[threadIdx.x + st];
    __syncthreads();
  }
  if (threadIdx.x == 0) ms[b] = red[0] / (float)TLEN;
}

// ---------------------------------------------------------------------------
// Memory-pool update: top-50 (stable argmax desc), filter > TAU1, project
// token -> 64-dim summary, insert in order. One block (64 threads) per batch.
// (With counts starting at 0 and S=POOL, the reference's replace branch is dead.)
// ---------------------------------------------------------------------------
__global__ __launch_bounds__(64) void mem_update_kernel(
    const float* __restrict__ scores, const float* __restrict__ y,
    const float* __restrict__ summ_w, float* __restrict__ pool,
    float* __restrict__ prio, int* __restrict__ counts) {
  const int b = blockIdx.x;
  const int tid = threadIdx.x;
  __shared__ float sc[TLEN];
  __shared__ float bv[64];
  __shared__ int   bi[64];
  __shared__ float selV[POOL_N];
  __shared__ int   selI[POOL_N];

  for (int i = tid; i < TLEN; i += 64) sc[i] = scores[b * TLEN + i];
  __syncthreads();

  for (int s = 0; s < POOL_N; ++s) {
    float best = -1e30f; int bidx = 1 << 30;
    for (int i = tid; i < TLEN; i += 64) {
      float v = sc[i];
      if (v > best) { best = v; bidx = i; }
    }
    bv[tid] = best; bi[tid] = bidx; __syncthreads();
    for (int st = 32; st > 0; st >>= 1) {
      if (tid < st) {
        if (bv[tid + st] > bv[tid] ||
            (bv[tid + st] == bv[tid] && bi[tid + st] < bi[tid])) {
          bv[tid] = bv[tid + st]; bi[tid] = bi[tid + st];
        }
      }
      __syncthreads();
    }
    if (tid == 0) { selV[s] = bv[0]; selI[s] = bi[0]; sc[bi[0]] = -1e30f; }
    __syncthreads();
  }

  int cnt = 0;
  for (int s = 0; s < POOL_N; ++s) {
    const float si = selV[s];
    if (si > TAU1F) {
      const float* tok = y + ((size_t)b * TLEN + selI[s]) * D_MODEL;
      const float* wr  = summ_w + (size_t)tid * D_MODEL;   // d = tid
      float acc = 0.f;
      for (int k = 0; k < D_MODEL; ++k) acc += tok[k] * wr[k];
      pool[((size_t)b * POOL_N + cnt) * SDIM + tid] = acc;
      if (tid == 0) prio[b * POOL_N + cnt] = si;
      ++cnt;
    }
  }
  for (int s = cnt; s < POOL_N; ++s) {
    pool[((size_t)b * POOL_N + s) * SDIM + tid] = 0.f;
    if (tid == 0) prio[b * POOL_N + s] = 0.f;
  }
  if (tid == 0) counts[b] = cnt;
}

// ---------------------------------------------------------------------------
// small projection: out[r,n] = in[r,:64] . w[n,:64]   (k_w / v_w on pool)
// ---------------------------------------------------------------------------
__global__ void small_proj_kernel(const float* __restrict__ in,
                                  const float* __restrict__ w,
                                  float* __restrict__ out, int rows, int N) {
  long long idx = (long long)blockIdx.x * blockDim.x + threadIdx.x;
  if (idx >= (long long)rows * N) return;
  const int r = (int)(idx / N), n = (int)(idx % N);
  const float* ir = in + (size_t)r * SDIM;
  const float* wr = w + (size_t)n * SDIM;
  float acc = 0.f;
#pragma unroll
  for (int i = 0; i < SDIM; ++i) acc += ir[i] * wr[i];
  out[idx] = acc;
}

// ---------------------------------------------------------------------------
// attention over 50 memory slots: one thread per (b, t) row
// ---------------------------------------------------------------------------
__global__ __launch_bounds__(256) void attn_kernel(
    const float* __restrict__ q, const float* __restrict__ k,
    const float* __restrict__ v, const int* __restrict__ counts,
    float* __restrict__ retr) {
  int row = blockIdx.x * blockDim.x + threadIdx.x;
  if (row >= BT) return;
  const int b = row / TLEN;
  const int cnt = counts[b];
  const float* qr = q + (size_t)row * SDIM;
  float logits[POOL_N];
  const float scale = 0.25f;    // 1/sqrt(HEAD_DIM_RET=16)
  float mx = -1e30f;
  for (int p = 0; p < POOL_N; ++p) {
    const float* kr = k + ((size_t)b * POOL_N + p) * SDIM;
    float acc = 0.f;
#pragma unroll
    for (int i = 0; i < SDIM; ++i) acc += qr[i] * kr[i];
    acc *= scale;
    if (p >= cnt) acc = -1e9f;
    logits[p] = acc;
    mx = fmaxf(mx, acc);
  }
  float sum = 0.f;
  for (int p = 0; p < POOL_N; ++p) { float e = expf(logits[p] - mx); logits[p] = e; sum += e; }
  const float rs = 1.f / sum;
  float* out = retr + (size_t)row * D_MODEL;
  for (int d = 0; d < D_MODEL; ++d) {
    float acc = 0.f;
    for (int p = 0; p < POOL_N; ++p)
      acc += logits[p] * v[((size_t)b * POOL_N + p) * D_MODEL + d];
    out[d] = acc * rs;
  }
}

// ---------------------------------------------------------------------------
// build bf16 concat([y, retrieved]) for the gate GEMM
// ---------------------------------------------------------------------------
__global__ void concat_bf16_kernel(const float* __restrict__ y,
                                   const float* __restrict__ retr,
                                   bf16_t* __restrict__ out) {
  long long idx = (long long)blockIdx.x * blockDim.x + threadIdx.x;
  if (idx >= (long long)BT * (2 * D_MODEL)) return;
  const int c = (int)(idx & (2 * D_MODEL - 1));
  const long long r = idx >> 11;
  float v = (c < D_MODEL) ? y[r * D_MODEL + c] : retr[r * D_MODEL + (c - D_MODEL)];
  out[idx] = (bf16_t)v;
}

// ---------------------------------------------------------------------------
// out = x + y + sigmoid(gate_lin) * retrieved * retrieve_mask[b]
// ---------------------------------------------------------------------------
__global__ void final_kernel(const float* __restrict__ x, const float* __restrict__ y,
                             const float* __restrict__ glin, const float* __restrict__ retr,
                             const float* __restrict__ ms, const int* __restrict__ counts,
                             float* __restrict__ out) {
  long long idx = (long long)blockIdx.x * blockDim.x + threadIdx.x;
  if (idx >= (long long)BT * D_MODEL) return;
  const int b = (int)(idx / ((long long)TLEN * D_MODEL));
  const float g = 1.f / (1.f + expf(-glin[idx]));
  const bool mask = (ms[b] > TAU2F) && (counts[b] > 0);
  out[idx] = x[idx] + y[idx] + (mask ? g * retr[idx] : 0.f);
}

// ---------------------------------------------------------------------------
// launch
// ---------------------------------------------------------------------------
static inline int cdiv_ll(long long a, int b) { return (int)((a + b - 1) / b); }

extern "C" void kernel_launch(void* const* d_in, const int* in_sizes, int n_in,
                              void* d_out, int out_size, void* d_ws, size_t ws_size,
                              hipStream_t stream) {
  const float* x       = (const float*)d_in[0];
  const float* norm_w  = (const float*)d_in[1];
  const float* in_w    = (const float*)d_in[2];
  const float* conv_w  = (const float*)d_in[3];
  const float* conv_b  = (const float*)d_in[4];
  const float* dt_bias = (const float*)d_in[5];
  const float* A_log   = (const float*)d_in[6];
  const float* D_param = (const float*)d_in[7];
  const float* gnorm_w = (const float*)d_in[8];
  const float* out_w   = (const float*)d_in[9];
  const float* sc_w1   = (const float*)d_in[10];
  const float* sc_w2   = (const float*)d_in[11];
  const float* summ_w  = (const float*)d_in[12];
  const float* q_w     = (const float*)d_in[13];
  const float* k_w     = (const float*)d_in[14];
  const float* v_w     = (const float*)d_in[15];
  const float* gate_w  = (const float*)d_in[16];
  float* outp = (float*)d_out;

  char* ws = (char*)d_ws;
  size_t off = 0;
  auto alloc = [&](size_t bytes) -> char* {
    char* p = ws + off;
    off = (off + bytes + 255) & ~(size_t)255;
    return p;
  };

  bf16_t* xn_bf   = (bf16_t*)alloc((size_t)BT * D_MODEL * 2);
  bf16_t* inw_bf  = (bf16_t*)alloc((size_t)D_IN_PROJ * D_MODEL * 2);
  float*  zxbcdt  = (float*)alloc((size_t)BT * D_IN_PROJ * 4);
  float*  xBC     = (float*)alloc((size_t)BT * CONV_DIM * 4);
  float*  dtb     = (float*)alloc((size_t)BT * NHEADS * 4);
  float*  dAb     = (float*)alloc((size_t)BT * NHEADS * 4);
  float*  yssm    = (float*)alloc((size_t)BT * D_INNER * 4);
  bf16_t* yg_bf   = (bf16_t*)alloc((size_t)BT * D_INNER * 2);
  bf16_t* outw_bf = (bf16_t*)alloc((size_t)D_MODEL * D_INNER * 2);
  float*  ybuf    = (float*)alloc((size_t)BT * D_MODEL * 4);
  bf16_t* y_bf    = (bf16_t*)alloc((size_t)BT * D_MODEL * 2);
  bf16_t* sw1_bf  = (bf16_t*)alloc((size_t)HID * D_MODEL * 2);
  float*  hbuf    = (float*)alloc((size_t)BT * HID * 4);
  float*  scores  = (float*)alloc((size_t)BT * 4);
  float*  ms      = (float*)alloc(BATCH * 4);
  float*  poolb   = (float*)alloc((size_t)BATCH * POOL_N * SDIM * 4);
  float*  priob   = (float*)alloc((size_t)BATCH * POOL_N * 4);
  int*    counts  = (int*)alloc(BATCH * 4);
  float*  kbuf    = (float*)alloc((size_t)BATCH * POOL_N * SDIM * 4);
  float*  vbuf    = (float*)alloc((size_t)BATCH * POOL_N * D_MODEL * 4);
  bf16_t* qw_bf   = (bf16_t*)alloc((size_t)SDIM * D_MODEL * 2);
  float*  qbuf    = (float*)alloc((size_t)BT * SDIM * 4);
  float*  retr    = (float*)alloc((size_t)BT * D_MODEL * 4);
  bf16_t* cat_bf  = (bf16_t*)alloc((size_t)BT * 2 * D_MODEL * 2);
  bf16_t* gw_bf   = (bf16_t*)alloc((size_t)D_MODEL * 2 * D_MODEL * 2);
  float*  glin    = (float*)alloc((size_t)BT * D_MODEL * 4);
  (void)ws_size; (void)in_sizes; (void)n_in; (void)out_size;

  auto gemm_blocks = [](int M, int N) { return ((M >> 5) * (N >> 5) + 7) / 8; };

  // 1. xn = rmsnorm(x) -> bf16
  rmsnorm_bf16_kernel<<<BT, 256, 0, stream>>>(x, norm_w, xn_bf, D_MODEL);

  // 2. zxbcdt = xn @ in_w^T  (WMMA)
  f32_to_bf16_kernel<<<cdiv_ll((long long)D_IN_PROJ * D_MODEL, 256), 256, 0, stream>>>(
      in_w, inw_bf, (long long)D_IN_PROJ * D_MODEL);
  gemm_bf16_wmma_kernel<<<gemm_blocks(BT, D_IN_PROJ), 256, 0, stream>>>(
      xn_bf, inw_bf, zxbcdt, BT, D_IN_PROJ, D_MODEL, 0);

  // 3. depthwise conv + silu, dt/dA
  conv_silu_kernel<<<cdiv_ll((long long)BT * CONV_DIM, 256), 256, 0, stream>>>(
      zxbcdt, conv_w, conv_b, xBC);
  dt_kernel<<<cdiv_ll((long long)BT * NHEADS, 256), 256, 0, stream>>>(
      zxbcdt, dt_bias, A_log, dtb, dAb);

  // 4. sequential SSM scan: one block per (batch, head)
  ssm_scan_kernel<<<BATCH * NHEADS, 256, 0, stream>>>(xBC, dtb, dAb, D_param, yssm);

  // 5. gated rmsnorm -> bf16, out projection (WMMA)
  gated_rmsnorm_kernel<<<BT, 256, 0, stream>>>(yssm, zxbcdt, gnorm_w, yg_bf);
  f32_to_bf16_kernel<<<cdiv_ll((long long)D_MODEL * D_INNER, 256), 256, 0, stream>>>(
      out_w, outw_bf, (long long)D_MODEL * D_INNER);
  gemm_bf16_wmma_kernel<<<gemm_blocks(BT, D_MODEL), 256, 0, stream>>>(
      yg_bf, outw_bf, ybuf, BT, D_MODEL, D_INNER, 0);
  f32_to_bf16_kernel<<<cdiv_ll((long long)BT * D_MODEL, 256), 256, 0, stream>>>(
      ybuf, y_bf, (long long)BT * D_MODEL);

  // 6. scorer MLP (WMMA + relu epilogue), sigmoid head, per-batch mean
  f32_to_bf16_kernel<<<cdiv_ll((long long)HID * D_MODEL, 256), 256, 0, stream>>>(
      sc_w1, sw1_bf, (long long)HID * D_MODEL);
  gemm_bf16_wmma_kernel<<<gemm_blocks(BT, HID), 256, 0, stream>>>(
      y_bf, sw1_bf, hbuf, BT, HID, D_MODEL, 1);
  score_kernel<<<cdiv_ll(BT, 256), 256, 0, stream>>>(hbuf, sc_w2, scores, BT);
  mean_kernel<<<BATCH, 256, 0, stream>>>(scores, ms);

  // 7. memory pool update (top-50, filter, summarize, insert)
  mem_update_kernel<<<BATCH, 64, 0, stream>>>(scores, ybuf, summ_w, poolb, priob, counts);

  // 8. k/v projections of the pool, q projection (WMMA)
  small_proj_kernel<<<cdiv_ll((long long)BATCH * POOL_N * SDIM, 256), 256, 0, stream>>>(
      poolb, k_w, kbuf, BATCH * POOL_N, SDIM);
  small_proj_kernel<<<cdiv_ll((long long)BATCH * POOL_N * D_MODEL, 256), 256, 0, stream>>>(
      poolb, v_w, vbuf, BATCH * POOL_N, D_MODEL);
  f32_to_bf16_kernel<<<cdiv_ll((long long)SDIM * D_MODEL, 256), 256, 0, stream>>>(
      q_w, qw_bf, (long long)SDIM * D_MODEL);
  gemm_bf16_wmma_kernel<<<gemm_blocks(BT, SDIM), 256, 0, stream>>>(
      y_bf, qw_bf, qbuf, BT, SDIM, D_MODEL, 0);

  // 9. retrieval attention over 50 slots
  attn_kernel<<<cdiv_ll(BT, 256), 256, 0, stream>>>(qbuf, kbuf, vbuf, counts, retr);

  // 10. gate GEMM on concat([y, retrieved]) (WMMA) and final combine
  concat_bf16_kernel<<<cdiv_ll((long long)BT * 2 * D_MODEL, 256), 256, 0, stream>>>(
      ybuf, retr, cat_bf);
  f32_to_bf16_kernel<<<cdiv_ll((long long)D_MODEL * 2 * D_MODEL, 256), 256, 0, stream>>>(
      gate_w, gw_bf, (long long)D_MODEL * 2 * D_MODEL);
  gemm_bf16_wmma_kernel<<<gemm_blocks(BT, D_MODEL), 256, 0, stream>>>(
      cat_bf, gw_bf, glin, BT, D_MODEL, 2 * D_MODEL, 0);
  final_kernel<<<cdiv_ll((long long)BT * D_MODEL, 256), 256, 0, stream>>>(
      x, ybuf, glin, retr, ms, counts, outp);
}
